// SAGE_Model_21887153341148
// MI455X (gfx1250) — matrix operand order repaired and checked by reading, not compiled, not run on hardware
//
#include <hip/hip_runtime.h>
#include <hip/hip_bf16.h>

typedef __attribute__((ext_vector_type(16))) _Float16 v16h;
typedef __attribute__((ext_vector_type(8)))  float    v8f;

#define NN 100000
#define NE 1600000

// ---------------- utility kernels ----------------

__global__ void zero_f32(float* __restrict__ p, int n) {
  int i = blockIdx.x * blockDim.x + threadIdx.x;
  if (i < n) p[i] = 0.0f;
}

__global__ void degree_kernel(const int* __restrict__ dst, float* __restrict__ cnt, int e) {
  int i = blockIdx.x * blockDim.x + threadIdx.x;
  if (i < e) atomicAdd(&cnt[dst[i]], 1.0f);
}

__global__ void recip_kernel(float* __restrict__ c, int n) {
  int i = blockIdx.x * blockDim.x + threadIdx.x;
  if (i < n) c[i] = 1.0f / fmaxf(c[i], 1.0f);
}

// pad x[N,13] -> xp[N,32] with zeros in cols 13..31
__global__ void pad13(const float* __restrict__ x, float* __restrict__ xp, int n32) {
  int i = blockIdx.x * blockDim.x + threadIdx.x;
  if (i >= n32) return;
  int r = i >> 5, c = i & 31;
  xp[i] = (c < 13) ? x[r * 13 + c] : 0.0f;
}

// Pack weight W[fin, fout] (f32, row-major) into per-lane WMMA B fragments (f16).
// Index: t = ((ct*kbCount + kb)*32 + lane)*16 + e
//   k = kb*32 + (e<8 ? (lane>>4)*8 + e : 16 + (lane>>4)*8 + (e-8)) ; col = ct*16 + (lane&15) ; zero-padded OOB.
__global__ void pack_w(const float* __restrict__ w, _Float16* __restrict__ pk,
                       int fin, int fout, int kbCount) {
  int t = blockIdx.x * blockDim.x + threadIdx.x;
  int total = 4 * kbCount * 32 * 16;
  if (t >= total) return;
  int e = t & 15;
  int lane = (t >> 4) & 31;
  int rest = t >> 9;
  int kb = rest % kbCount;
  int ct = rest / kbCount;
  int k = kb * 32 + (e < 8 ? (lane >> 4) * 8 + e : 16 + (lane >> 4) * 8 + (e - 8));
  int col = ct * 16 + (lane & 15);
  pk[t] = (k < fin && col < fout) ? (_Float16)w[k * fout + col] : (_Float16)0.0f;
}

// ---------------- scatter (segment-sum) kernels ----------------

// layer-0 scatter: x[N,13] -> agg (stride 32, cols 13..31 stay zero)
__global__ void scatter13(const float* __restrict__ x, const int* __restrict__ src,
                          const int* __restrict__ dst, float* __restrict__ agg, int e) {
  int i = blockIdx.x * blockDim.x + threadIdx.x;
  if (i >= e) return;
  const float* xp = x + (size_t)src[i] * 13;
  float* ap = agg + (size_t)dst[i] * 32;
#pragma unroll
  for (int f = 0; f < 13; ++f) atomicAdd(&ap[f], xp[f]);
}

// one thread handles one float4 chunk of one edge (16 chunks/edge, F=64)
__global__ void scatter64(const float* __restrict__ h, const int* __restrict__ src,
                          const int* __restrict__ dst, float* __restrict__ agg, int e) {
  int t = blockIdx.x * blockDim.x + threadIdx.x;
  int i = t >> 4;
  if (i >= e) return;
  int c = (t & 15) << 2;
  float4 v = *(const float4*)(h + (size_t)src[i] * 64 + c);
  float* ap = agg + (size_t)dst[i] * 64 + c;
  atomicAdd(&ap[0], v.x);
  atomicAdd(&ap[1], v.y);
  atomicAdd(&ap[2], v.z);
  atomicAdd(&ap[3], v.w);
}

// ---------------- WMMA SAGE layer ----------------
// out[row, col] = act( (agg[row,:]*inv[row]) @ wl + hin[row,:] @ wr + bias[col] )
// Block = 256 threads = 8 waves; block tile = 32 rows x 64 cols; wave tile = 16x16.
// A(16x32 f16): lane L -> row M=L&15; elems 0..7 = K (L>>4)*8.., elems 8..15 = +16.
// B: pre-packed per-lane fragments (one 32B load each).
// C/D(16x16 f32): lane L -> col N=L&15; vgpr r -> row (L>>4)*8 + r.

__device__ __forceinline__ v16h ldA(const float* __restrict__ rowp, int kb, int ks,
                                    float s) {
  const float4 a0 = *(const float4*)(rowp + kb + ks * 8);
  const float4 a1 = *(const float4*)(rowp + kb + ks * 8 + 4);
  const float4 a2 = *(const float4*)(rowp + kb + 16 + ks * 8);
  const float4 a3 = *(const float4*)(rowp + kb + 16 + ks * 8 + 4);
  v16h a;
  a[0]  = (_Float16)(a0.x * s); a[1]  = (_Float16)(a0.y * s);
  a[2]  = (_Float16)(a0.z * s); a[3]  = (_Float16)(a0.w * s);
  a[4]  = (_Float16)(a1.x * s); a[5]  = (_Float16)(a1.y * s);
  a[6]  = (_Float16)(a1.z * s); a[7]  = (_Float16)(a1.w * s);
  a[8]  = (_Float16)(a2.x * s); a[9]  = (_Float16)(a2.y * s);
  a[10] = (_Float16)(a2.z * s); a[11] = (_Float16)(a2.w * s);
  a[12] = (_Float16)(a3.x * s); a[13] = (_Float16)(a3.y * s);
  a[14] = (_Float16)(a3.z * s); a[15] = (_Float16)(a3.w * s);
  return a;
}

template <int FIN, int FOUT, int ACT>
__global__ __launch_bounds__(256) void sage_gemm(
    const float* __restrict__ agg, const float* __restrict__ inv,
    const float* __restrict__ hin, const _Float16* __restrict__ wlp,
    const _Float16* __restrict__ wrp, const float* __restrict__ bias,
    float* __restrict__ out) {
  constexpr int KB = FIN / 32;
  const int lane = threadIdx.x & 31;
  const int wave = threadIdx.x >> 5;
  const int rowBase = blockIdx.x * 32 + (wave >> 2) * 16;
  const int ct = wave & 3;
  const int m  = lane & 15;
  const int ks = lane >> 4;
  const int row = rowBase + m;
  const int cc  = ct * 16 + m;

  v8f c = {0.f, 0.f, 0.f, 0.f, 0.f, 0.f, 0.f, 0.f};

  const float scale = inv[row];
  const float* aggRow = agg + (size_t)row * FIN;
  const float* hinRow = hin + (size_t)row * FIN;
  const _Float16* wlL = wlp + ((size_t)(ct * KB) * 32 + lane) * 16;
  const _Float16* wrL = wrp + ((size_t)(ct * KB) * 32 + lane) * 16;

#pragma unroll
  for (int kbi = 0; kbi < KB; ++kbi) {
    v16h a = ldA(aggRow, kbi * 32, ks, scale);
    v16h b = *(const v16h*)(wlL + (size_t)kbi * 32 * 16);
    c = __builtin_amdgcn_wmma_f32_16x16x32_f16(false, a, false, b, (short)0, c,
                                               false, false);
  }
#pragma unroll
  for (int kbi = 0; kbi < KB; ++kbi) {
    v16h a = ldA(hinRow, kbi * 32, ks, 1.0f);
    v16h b = *(const v16h*)(wrL + (size_t)kbi * 32 * 16);
    c = __builtin_amdgcn_wmma_f32_16x16x32_f16(false, a, false, b, (short)0, c,
                                               false, false);
  }

  const float bv = (cc < FOUT) ? bias[cc] : 0.0f;
#pragma unroll
  for (int r = 0; r < 8; ++r) {
    int mrow = rowBase + ks * 8 + r;
    float v = c[r] + bv;
    if (ACT == 0)
      v = fmaxf(v, 0.0f);
    else
      v = 1.0f / (1.0f + __expf(-v));
    if (cc < FOUT) out[(size_t)mrow * FOUT + cc] = v;
  }
}

// ---------------- launch ----------------

extern "C" void kernel_launch(void* const* d_in, const int* in_sizes, int n_in,
                              void* d_out, int out_size, void* d_ws, size_t ws_size,
                              hipStream_t stream) {
  (void)in_sizes; (void)n_in; (void)out_size; (void)ws_size;

  const float* x  = (const float*)d_in[0];
  const int*   ei = (const int*)d_in[1];
  const int*   src = ei;
  const int*   dst = ei + NE;

  const float* wl[5] = {(const float*)d_in[2],  (const float*)d_in[5],
                        (const float*)d_in[8],  (const float*)d_in[11],
                        (const float*)d_in[14]};
  const float* wr[5] = {(const float*)d_in[3],  (const float*)d_in[6],
                        (const float*)d_in[9],  (const float*)d_in[12],
                        (const float*)d_in[15]};
  const float* bs[5] = {(const float*)d_in[4],  (const float*)d_in[7],
                        (const float*)d_in[10], (const float*)d_in[13],
                        (const float*)d_in[16]};

  // workspace layout (floats)
  float* hA  = (float*)d_ws;           // N*64
  float* hB  = hA + (size_t)NN * 64;   // N*64
  float* agg = hB + (size_t)NN * 64;   // N*64 (layer 0 uses stride-32 region)
  float* inv = agg + (size_t)NN * 64;  // N
  float* xp  = inv + NN;               // N*32 (padded x)
  _Float16* wh = (_Float16*)(xp + (size_t)NN * 32);
  // packed weight fragments: size = 4*KB*32*16 halves
  const int P1 = 4 * 1 * 32 * 16;  // KB=1 (layer 0)
  const int P2 = 4 * 2 * 32 * 16;  // KB=2 (layers 1..4)
  _Float16* wl0p = wh;
  _Float16* wr0p = wl0p + P1;
  _Float16* wl1p = wr0p + P1;
  _Float16* wr1p = wl1p + P2;
  _Float16* wl2p = wr1p + P2;
  _Float16* wr2p = wl2p + P2;
  _Float16* wl3p = wr2p + P2;
  _Float16* wr3p = wl3p + P2;
  _Float16* wl4p = wr3p + P2;
  _Float16* wr4p = wl4p + P2;

  const int T = 256;
  auto gb = [](long long n, int t) { return (int)((n + t - 1) / t); };

  // pack weights into WMMA B-fragment layout (zero-padded)
  pack_w<<<gb(P1, T), T, 0, stream>>>(wl[0], wl0p, 13, 64, 1);
  pack_w<<<gb(P1, T), T, 0, stream>>>(wr[0], wr0p, 13, 64, 1);
  pack_w<<<gb(P2, T), T, 0, stream>>>(wl[1], wl1p, 64, 64, 2);
  pack_w<<<gb(P2, T), T, 0, stream>>>(wr[1], wr1p, 64, 64, 2);
  pack_w<<<gb(P2, T), T, 0, stream>>>(wl[2], wl2p, 64, 64, 2);
  pack_w<<<gb(P2, T), T, 0, stream>>>(wr[2], wr2p, 64, 64, 2);
  pack_w<<<gb(P2, T), T, 0, stream>>>(wl[3], wl3p, 64, 64, 2);
  pack_w<<<gb(P2, T), T, 0, stream>>>(wr[3], wr3p, 64, 64, 2);
  pack_w<<<gb(P2, T), T, 0, stream>>>(wl[4], wl4p, 64, 1, 2);
  pack_w<<<gb(P2, T), T, 0, stream>>>(wr[4], wr4p, 64, 1, 2);

  // padded x
  pad13<<<gb((long long)NN * 32, T), T, 0, stream>>>(x, xp, NN * 32);

  // inverse mean degree (graph-constant across layers)
  zero_f32<<<gb(NN, T), T, 0, stream>>>(inv, NN);
  degree_kernel<<<gb(NE, T), T, 0, stream>>>(dst, inv, NE);
  recip_kernel<<<gb(NN, T), T, 0, stream>>>(inv, NN);

  const int GEMM_BLOCKS = NN / 32;  // 3125, exact
  const int SC64_BLOCKS = gb((long long)NE * 16, T);

  // layer 0: xp[N,32] -> hA[N,64]
  zero_f32<<<gb((long long)NN * 32, T), T, 0, stream>>>(agg, NN * 32);
  scatter13<<<gb(NE, T), T, 0, stream>>>(x, src, dst, agg, NE);
  sage_gemm<32, 64, 0><<<GEMM_BLOCKS, T, 0, stream>>>(agg, inv, xp, wl0p, wr0p, bs[0], hA);

  // layer 1: hA -> hB
  zero_f32<<<gb((long long)NN * 64, T), T, 0, stream>>>(agg, NN * 64);
  scatter64<<<SC64_BLOCKS, T, 0, stream>>>(hA, src, dst, agg, NE);
  sage_gemm<64, 64, 0><<<GEMM_BLOCKS, T, 0, stream>>>(agg, inv, hA, wl1p, wr1p, bs[1], hB);

  // layer 2: hB -> hA
  zero_f32<<<gb((long long)NN * 64, T), T, 0, stream>>>(agg, NN * 64);
  scatter64<<<SC64_BLOCKS, T, 0, stream>>>(hB, src, dst, agg, NE);
  sage_gemm<64, 64, 0><<<GEMM_BLOCKS, T, 0, stream>>>(agg, inv, hB, wl2p, wr2p, bs[2], hA);

  // layer 3: hA -> hB
  zero_f32<<<gb((long long)NN * 64, T), T, 0, stream>>>(agg, NN * 64);
  scatter64<<<SC64_BLOCKS, T, 0, stream>>>(hA, src, dst, agg, NE);
  sage_gemm<64, 64, 0><<<GEMM_BLOCKS, T, 0, stream>>>(agg, inv, hA, wl3p, wr3p, bs[3], hB);

  // layer 4: hB -> d_out (sigmoid, F_out=1)
  zero_f32<<<gb((long long)NN * 64, T), T, 0, stream>>>(agg, NN * 64);
  scatter64<<<SC64_BLOCKS, T, 0, stream>>>(hB, src, dst, agg, NE);
  sage_gemm<64, 1, 1><<<GEMM_BLOCKS, T, 0, stream>>>(agg, inv, hB, wl4p, wr4p, bs[4],
                                                     (float*)d_out);
}
